// HatNet_55508157333679
// MI455X (gfx1250) — matrix operand order repaired and checked by reading, not compiled, not run on hardware
//
#include <hip/hip_runtime.h>

// ---------------------------------------------------------------------------
// MI455X (gfx1250, wave32) implementation of the HatNet forward pass.
// All GEMMs go through v_wmma_f32_16x16x32_bf16 (bf16 A/B, f32 accumulate),
// with fragment-ready swizzled LDS tiles (wide ds_load_b128 operand reads),
// register double-buffering of the next K tile, and an unchecked fast path
// for interior blocks / full K tiles.
// ---------------------------------------------------------------------------

typedef __attribute__((ext_vector_type(16))) __bf16 v16bf;
typedef __attribute__((ext_vector_type(4)))  __bf16 v4bf;
typedef __attribute__((ext_vector_type(8)))  float  v8f;

#define GSL(i, n) for (long long i = (long long)blockIdx.x * blockDim.x + threadIdx.x; \
                       i < (n); i += (long long)gridDim.x * blockDim.x)

// ----------------------------- elementwise ---------------------------------
__global__ void k_fill(float* p, long long n, float v) { GSL(i, n) p[i] = v; }
__global__ void k_copy(float* d, const float* s, long long n) { GSL(i, n) d[i] = s[i]; }

// GIN sparse aggregation: acc[dst] += x[src] over all edges (atomic scatter)
__global__ void k_edge_scatter(const int* __restrict__ ei, long long E,
                               const float* __restrict__ x, float* acc, int D) {
  long long total = E * (long long)D;
  GSL(i, total) {
    long long e = i / D; int c = (int)(i - e * D);
    int s = ei[e], d = ei[E + e];
    if (c == 0) __builtin_prefetch(&x[(long long)s * D], 0, 0);   // global_prefetch_b8
    atomicAdd(&acc[(long long)d * D + c], x[(long long)s * D + c]);
  }
}

// dense adjacency build: adj[b, src%N, dst%N] += 1
__global__ void k_adj_scatter(const int* __restrict__ ei, long long E, float* adj) {
  GSL(e, E) {
    int s = ei[e], d = ei[E + e];
    int b = s / 548;
    atomicAdd(&adj[(long long)b * 548 * 548 + (long long)(s % 548) * 548 + (d % 548)], 1.0f);
  }
}

// ------------------------------ WMMA GEMM ----------------------------------
// C[b] = act( op(A[b]) (MxK) @ B[b] (KxN) + bias + Cadd[b] )
// 256-thread block = 8 waves; each wave owns one 16x16 WMMA tile of a
// 32(M) x 64(N) block tile; K stepped by 32.
// LDS tiles are stored in WMMA-fragment order (ISA 7.12.2), so each lane's
// operand is one contiguous 32-byte LDS read; the fp32->bf16 converting
// loaders write packed 4x-bf16 quads (ds_store_b64). The next K tile's global
// loads are double-buffered in registers to overlap with the WMMA. Interior
// blocks take a bounds-check-free fast path (checked loads only on the final
// partial K tile); only edge blocks run the fully predicated loader.
#define TM 32
#define TN 64
#define TK 32

__global__ __launch_bounds__(256)
void k_gemm(const float* __restrict__ A, long long sA, int lda, int transA,
            const float* __restrict__ Bm, long long sB, int ldb,
            const float* __restrict__ bias,
            const float* __restrict__ Cadd, long long sCa, int ldca,
            float* __restrict__ C, long long sC, int ldc,
            int M, int Nn, int K, int relu)
{
  // fragment-order LDS: [tile group][lane][slot]
  __shared__ alignas(32) __bf16 AsF[2][32][16];   // 2 KB
  __shared__ alignas(32) __bf16 BsF[4][32][16];   // 4 KB

  const int bz = blockIdx.z;
  const float* Ab = A  + (long long)bz * sA;
  const float* Bb = Bm + (long long)bz * sB;
  float*       Cb = C  + (long long)bz * sC;

  const int m0 = blockIdx.y * TM;
  const int n0 = blockIdx.x * TN;
  const int tid  = threadIdx.x;
  const int wave = tid >> 5, lane = tid & 31;
  const int wm = wave >> 2, wn = wave & 3;       // 2 x 4 wave grid
  const int half = lane >> 4, l16 = lane & 15;

  // --- per-thread loader decomposition (constant over K loop) ---
  // A: one k-quad per thread: element (a_m, a_k..a_k+3)
  const int a_m = tid >> 3;             // 0..31
  const int a_k = (tid & 7) * 4;        // 0..28 step 4
  const int a_g = a_m >> 4;
  const int a_lane = ((a_k >> 3) & 1) * 16 + (a_m & 15);
  const int a_slot = ((a_k >> 4) << 3) | (a_k & 7);
  // B: two k-quads per thread: element (b_k..b_k+3, b_n)
  int b_n[2], b_k[2], b_g[2], b_lane[2], b_slot[2];
  #pragma unroll
  for (int j = 0; j < 2; ++j) {
    int qid = tid + j * 256;
    b_n[j] = qid & 63;
    b_k[j] = (qid >> 6) * 4;
    b_g[j] = b_n[j] >> 4;
    b_lane[j] = ((b_k[j] >> 3) & 1) * 16 + (b_n[j] & 15);
    b_slot[j] = ((b_k[j] >> 4) << 3) | (b_k[j] & 7);
  }
  const int gm_a = m0 + a_m;

  float aq[4], bq[2][4];

  // unchecked loads (interior block, full K tile)
  auto load_fast = [&](int k0) {
    #pragma unroll
    for (int q = 0; q < 4; ++q) {
      int gk = k0 + a_k + q;
      aq[q] = transA ? Ab[(long long)gk * lda + gm_a] : Ab[(long long)gm_a * lda + gk];
    }
    #pragma unroll
    for (int j = 0; j < 2; ++j) {
      long long gn = n0 + b_n[j];
      #pragma unroll
      for (int q = 0; q < 4; ++q)
        bq[j][q] = Bb[(long long)(k0 + b_k[j] + q) * ldb + gn];
    }
  };
  // fully predicated loads (edge blocks / partial K tile)
  auto load_safe = [&](int k0) {
    #pragma unroll
    for (int q = 0; q < 4; ++q) {
      int gk = k0 + a_k + q;
      float vv = 0.f;
      if (gm_a < M && gk < K)
        vv = transA ? Ab[(long long)gk * lda + gm_a] : Ab[(long long)gm_a * lda + gk];
      aq[q] = vv;
    }
    #pragma unroll
    for (int j = 0; j < 2; ++j) {
      int gn = n0 + b_n[j];
      #pragma unroll
      for (int q = 0; q < 4; ++q) {
        int gk = k0 + b_k[j] + q;
        bq[j][q] = (gk < K && gn < Nn) ? Bb[(long long)gk * ldb + gn] : 0.f;
      }
    }
  };

  auto store_tiles = [&]() {
    v4bf pa;
    #pragma unroll
    for (int q = 0; q < 4; ++q) pa[q] = (__bf16)aq[q];
    *(v4bf*)&AsF[a_g][a_lane][a_slot] = pa;
    #pragma unroll
    for (int j = 0; j < 2; ++j) {
      v4bf pb;
      #pragma unroll
      for (int q = 0; q < 4; ++q) pb[q] = (__bf16)bq[j][q];
      *(v4bf*)&BsF[b_g[j]][b_lane[j]][b_slot[j]] = pb;
    }
  };

  v8f acc = {};
  auto mma = [&]() {
    v16bf af  = *(const v16bf*)&AsF[wm][lane][0];
    v16bf bfr = *(const v16bf*)&BsF[wn][lane][0];
    acc = __builtin_amdgcn_wmma_f32_16x16x32_bf16(
        /*neg_a=*/false, af, /*neg_b=*/false, bfr,
        /*c_mod=*/(short)0, acc, /*reuse_a=*/false, /*reuse_b=*/false);
  };

  const bool interior = (m0 + TM <= M) && (n0 + TN <= Nn);

  if (interior) {
    const int Kmain = (K / TK) * TK;          // full tiles
    if (Kmain > 0) load_fast(0); else load_safe(0);
    int k0 = 0;
    for (; k0 < Kmain; k0 += TK) {
      __syncthreads();                        // previous fragment reads done
      store_tiles();
      __syncthreads();
      int kn = k0 + TK;
      if (kn < Kmain)      load_fast(kn);     // hot: no bounds checks
      else if (kn < K)     load_safe(kn);     // k-tail tile
      mma();
    }
    if (k0 < K) {                             // partial K tile
      __syncthreads();
      store_tiles();
      __syncthreads();
      mma();
    }
  } else {
    load_safe(0);
    for (int k0 = 0; k0 < K; k0 += TK) {
      __syncthreads();
      store_tiles();
      __syncthreads();
      if (k0 + TK < K) load_safe(k0 + TK);
      mma();
    }
  }

  // epilogue per C/D f32 layout: VGPR r -> row half*8 + r, col = lane&15
  const int col = n0 + wn * 16 + l16;
  #pragma unroll
  for (int r = 0; r < 8; ++r) {
    int row = m0 + wm * 16 + half * 8 + r;
    if (row < M && col < Nn) {
      float vv = acc[r];
      if (bias) vv += bias[col];
      if (Cadd) vv += Cadd[(long long)bz * sCa + (long long)row * ldca + col];
      if (relu) vv = vv > 0.f ? vv : 0.f;
      Cb[(long long)row * ldc + col] = vv;
    }
  }
}

// ------------------------- reductions / pointwise --------------------------
__global__ void k_bn_stats(const float* __restrict__ h, int M, int N,
                           float* mean, float* rstd) {
  int c = blockIdx.x, t = threadIdx.x;
  __shared__ float red[256];
  float s = 0.f, s2 = 0.f;
  for (int r = t; r < M; r += 256) { float v = h[(long long)r * N + c]; s += v; s2 += v * v; }
  red[t] = s; __syncthreads();
  for (int st = 128; st > 0; st >>= 1) { if (t < st) red[t] += red[t + st]; __syncthreads(); }
  float sm = red[0]; __syncthreads();
  red[t] = s2; __syncthreads();
  for (int st = 128; st > 0; st >>= 1) { if (t < st) red[t] += red[t + st]; __syncthreads(); }
  if (t == 0) {
    float mu = sm / M;
    float var = red[0] / M - mu * mu;
    mean[c] = mu; rstd[c] = rsqrtf(var + 1e-5f);
  }
}

__global__ void k_bn_relu(float* h, const float* mean, const float* rstd,
                          const float* g, const float* be, long long n, int N) {
  GSL(i, n) {
    int c = (int)(i % N);
    float v = (h[i] - mean[c]) * rstd[c] * g[c] + be[c];
    h[i] = v > 0.f ? v : 0.f;
  }
}

__global__ void k_softmax(float* s, int K) {
  long long row = blockIdx.x;
  float* p = s + row * (long long)K;
  __shared__ float red[256];
  int t = threadIdx.x;
  float mx = -3.4e38f;
  for (int k = t; k < K; k += 256) mx = fmaxf(mx, p[k]);
  red[t] = mx; __syncthreads();
  for (int st = 128; st > 0; st >>= 1) { if (t < st) red[t] = fmaxf(red[t], red[t + st]); __syncthreads(); }
  mx = red[0]; __syncthreads();
  float sum = 0.f;
  for (int k = t; k < K; k += 256) sum += __expf(p[k] - mx);
  red[t] = sum; __syncthreads();
  for (int st = 128; st > 0; st >>= 1) { if (t < st) red[t] += red[t + st]; __syncthreads(); }
  float inv = 1.f / red[0]; __syncthreads();
  for (int k = t; k < K; k += 256) p[k] = __expf(p[k] - mx) * inv;
}

// mode 0: out[row] = sum_k m[row,k] ; mode 1: sqrt(sum)+EPS (mincut degree norm)
__global__ void k_rowsum(const float* __restrict__ m, int K, float* out, int mode) {
  long long row = blockIdx.x;
  const float* p = m + row * (long long)K;
  __shared__ float red[256];
  int t = threadIdx.x;
  float s = 0.f;
  for (int k = t; k < K; k += 256) s += p[k];
  red[t] = s; __syncthreads();
  for (int st = 128; st > 0; st >>= 1) { if (t < st) red[t] += red[t + st]; __syncthreads(); }
  if (t == 0) out[row] = mode ? (sqrtf(red[0]) + 1e-15f) : red[0];
}

__global__ void k_trace(const float* __restrict__ m, int K, float* out) {
  int b = blockIdx.x, t = threadIdx.x;
  const float* p = m + (long long)b * K * K;
  __shared__ float red[256];
  float s = 0.f;
  for (int k = t; k < K; k += 256) s += p[(long long)k * K + k];
  red[t] = s; __syncthreads();
  for (int st = 128; st > 0; st >>= 1) { if (t < st) red[t] += red[t + st]; __syncthreads(); }
  if (t == 0) out[b] = red[0];
}

// den[b] = sum_{n,k} s[b,n,k]^2 * deg[b,n]
__global__ void k_den(const float* __restrict__ s, const float* __restrict__ deg,
                      int Nr, int K, float* den) {
  int b = blockIdx.x, t = threadIdx.x;
  const float* sp = s + (long long)b * Nr * K;
  const float* dp = deg + (long long)b * Nr;
  __shared__ float red[256];
  float acc = 0.f;
  long long tot = (long long)Nr * K;
  for (long long i = t; i < tot; i += 256) {
    int n = (int)(i / K);
    float v = sp[i];
    acc += v * v * dp[n];
  }
  red[t] = acc; __syncthreads();
  for (int st = 128; st > 0; st >>= 1) { if (t < st) red[t] += red[t + st]; __syncthreads(); }
  if (t == 0) den[b] = red[0];
}

// ortho[b] = || ss/||ss||_F - I/sqrt(K) ||_F
__global__ void k_ortho(const float* __restrict__ ss, int K, float* o) {
  int b = blockIdx.x, t = threadIdx.x;
  const float* p = ss + (long long)b * K * K;
  __shared__ float red[256];
  long long KK = (long long)K * K;
  float s2 = 0.f;
  for (long long i = t; i < KK; i += 256) { float v = p[i]; s2 += v * v; }
  red[t] = s2; __syncthreads();
  for (int st = 128; st > 0; st >>= 1) { if (t < st) red[t] += red[t + st]; __syncthreads(); }
  float invn = 1.f / sqrtf(red[0]); __syncthreads();
  float diag = rsqrtf((float)K);
  float acc = 0.f;
  for (long long i = t; i < KK; i += 256) {
    int r = (int)(i / K), c = (int)(i % K);
    float v = p[i] * invn - ((r == c) ? diag : 0.f);
    acc += v * v;
  }
  red[t] = acc; __syncthreads();
  for (int st = 128; st > 0; st >>= 1) { if (t < st) red[t] += red[t + st]; __syncthreads(); }
  if (t == 0) o[b] = sqrtf(red[0]);
}

__global__ void k_zero_diag(float* m, int K) {
  GSL(i, (long long)128 * K) {
    long long b = i / K, k = i % K;
    m[b * (long long)K * K + k * (long long)K + k] = 0.f;
  }
}

__global__ void k_adj_norm(float* m, const float* __restrict__ dv, int K) {
  GSL(i, (long long)128 * K * K) {
    long long b = i / ((long long)K * K);
    long long r = (i / K) % K, c = i % K;
    m[i] = m[i] / (dv[b * K + r] * dv[b * K + c]);
  }
}

// pooled sum/mean/max over M rows of [B, M, 128] into h[B, 2688] slices
__global__ void k_pool(const float* __restrict__ xd, int M, float* h,
                       int off_sum, int off_mean, int off_max) {
  int d = blockIdx.x, b = blockIdx.y, t = threadIdx.x;
  const float* p = xd + ((long long)b * M) * 128 + d;
  __shared__ float rs[256], rm[256];
  float s = 0.f, mx = -3.4e38f;
  for (int r = t; r < M; r += 256) { float v = p[(long long)r * 128]; s += v; mx = fmaxf(mx, v); }
  rs[t] = s; rm[t] = mx; __syncthreads();
  for (int st = 128; st > 0; st >>= 1) {
    if (t < st) { rs[t] += rs[t + st]; rm[t] = fmaxf(rm[t], rm[t + st]); }
    __syncthreads();
  }
  if (t == 0) {
    float* hb = h + (long long)b * 2688;
    hb[off_sum  + d] = rs[0];
    hb[off_mean + d] = rs[0] / M;
    hb[off_max  + d] = rm[0];
  }
}

__global__ void k_losses(const float* num1, const float* den1, const float* o1,
                         const float* num2, const float* den2, const float* o2,
                         float* out) {
  __shared__ float r1[128], r2[128];
  int t = threadIdx.x;   // 128 threads == B
  r1[t] = -(num1[t] / den1[t]) - (num2[t] / den2[t]);
  r2[t] = o1[t] + o2[t];
  __syncthreads();
  for (int st = 64; st > 0; st >>= 1) {
    if (t < st) { r1[t] += r1[t + st]; r2[t] += r2[t + st]; }
    __syncthreads();
  }
  if (t == 0) { out[512] = r1[0] / 128.f; out[513] = r2[0] / 128.f; }
}

// ------------------------------ host side ----------------------------------
static inline int gf(long long n) {
  long long b = (n + 255) / 256;
  return (int)(b > 32768 ? 32768 : (b < 1 ? 1 : b));
}

static inline void launch_gemm(hipStream_t st,
                               const float* A, long long sA, int lda, int tA,
                               const float* B, long long sB, int ldb,
                               const float* bias,
                               const float* Cadd, long long sCa, int ldca,
                               float* C, long long sC, int ldc,
                               int M, int N, int K, int relu, int batch) {
  dim3 g((N + TN - 1) / TN, (M + TM - 1) / TM, batch);
  k_gemm<<<g, 256, 0, st>>>(A, sA, lda, tA, B, sB, ldb, bias,
                            Cadd, sCa, ldca, C, sC, ldc, M, N, K, relu);
}

struct NNp { const float *w1, *b1, *g, *be, *w2, *b2; };

static inline void nn_block(hipStream_t st, const float* in, int M, int Kin,
                            const NNp& p, float* htmp, float* out,
                            float* bnm, float* bnr) {
  launch_gemm(st, in, 0, Kin, 0, p.w1, 0, 128, p.b1, nullptr, 0, 0,
              htmp, 0, 128, M, 128, Kin, 0, 1);
  k_bn_stats<<<128, 256, 0, st>>>(htmp, M, 128, bnm, bnr);
  long long tot = (long long)M * 128;
  k_bn_relu<<<gf(tot), 256, 0, st>>>(htmp, bnm, bnr, p.g, p.be, tot, 128);
  launch_gemm(st, htmp, 0, 128, 0, p.w2, 0, 128, p.b2, nullptr, 0, 0,
              out, 0, 128, M, 128, 128, 1, 1);
}

extern "C" void kernel_launch(void* const* d_in, const int* in_sizes, int n_in,
                              void* d_out, int out_size, void* d_ws, size_t ws_size,
                              hipStream_t stream) {
  (void)in_sizes; (void)n_in; (void)out_size; (void)ws_size;

  const int Bx = 128, Nn = 548, D = 128, INC = 16, K1 = 274, K2 = 137, Hd = 672, Fd = 2688;
  const long long Etot = (long long)Bx * 17536;           // 2,244,608 edges
  const long long R1 = (long long)Bx * Nn;                // 70,144 node rows
  const long long R2 = (long long)Bx * K1;                // 35,072
  const long long R3 = (long long)Bx * K2;                // 17,536

  const float* x  = (const float*)d_in[0];
  const int*   ei = (const int*)d_in[1];
  // d_in[2] = batch (implicit: contiguous graphs of N nodes)
  int pi = 3;
  NNp c[5];
  for (int i = 0; i < 5; ++i) {
    c[i].w1 = (const float*)d_in[pi + 0]; c[i].b1 = (const float*)d_in[pi + 1];
    c[i].g  = (const float*)d_in[pi + 2]; c[i].be = (const float*)d_in[pi + 3];
    c[i].w2 = (const float*)d_in[pi + 4]; c[i].b2 = (const float*)d_in[pi + 5];
    pi += 6;
  }
  const float* p1w = (const float*)d_in[33]; const float* p1b = (const float*)d_in[34];
  const float* p2w = (const float*)d_in[35]; const float* p2b = (const float*)d_in[36];
  const float* f1w = (const float*)d_in[37]; const float* f1b = (const float*)d_in[38];
  const float* f1g = (const float*)d_in[39]; const float* f1be = (const float*)d_in[40];
  const float* f2w = (const float*)d_in[41]; const float* f2b = (const float*)d_in[42];
  float* out = (float*)d_out;

  // ---- workspace bump allocation (floats) ----
  float* W = (float*)d_ws; size_t off = 0;
  auto A_ = [&](long long n) { float* p = W + off; off += (size_t)n; return p; };
  float* t1    = A_(R1 * INC);
  float* htmp  = A_(R1 * D);
  float* x1    = A_(R1 * D);
  float* t2    = A_(R1 * D);
  float* x2    = A_(R1 * D);
  float* s1    = A_(R1 * K1);
  float* xp1   = A_((long long)Bx * K1 * D);
  float* sta1  = A_((long long)Bx * K1 * Nn);   // also reused as ss1
  float* oadj1 = A_((long long)Bx * K1 * K1);
  float* deg1  = A_(R1);
  float* dvec1 = A_(R2);
  float* num1 = A_(Bx); float* den1 = A_(Bx); float* o1 = A_(Bx);
  float* num2 = A_(Bx); float* den2 = A_(Bx); float* o2 = A_(Bx);
  float* hcat = A_((long long)Bx * Fd);
  float* h1   = A_((long long)Bx * Hd);
  float* bnm  = A_(Hd); float* bnr = A_(Hd);
  float* deg2 = A_(R2); float* dvec2 = A_(R3);
  float* adj  = A_((long long)Bx * Nn * Nn);    // 38.4M floats, dead after mincut-1
  // overlay post-mincut-1 temporaries onto the adjacency region
  float* tsh   = adj;                                 // residual GIN inputs t3/t4/t5
  float* x3    = tsh  + (long long)Bx * K1 * D;
  float* x4    = x3   + (long long)Bx * K1 * D;
  float* s2    = x4   + (long long)Bx * K1 * D;
  float* xp2   = s2   + (long long)Bx * K1 * K2;
  float* sta2  = xp2  + (long long)Bx * K2 * D;       // also reused as ss2
  float* oadj2 = sta2 + (long long)Bx * K2 * K1;
  float* x5    = x3;                                  // x3 dead once x4 exists

  // ---- sparse GIN layer 1: nn(x + scatter_sum(x[src] -> dst)) ----
  k_copy<<<gf(R1 * INC), 256, 0, stream>>>(t1, x, R1 * INC);
  k_edge_scatter<<<gf(Etot * INC), 256, 0, stream>>>(ei, Etot, x, t1, INC);
  nn_block(stream, t1, (int)R1, INC, c[0], htmp, x1, bnm, bnr);
  k_pool<<<dim3(D, Bx), 256, 0, stream>>>(x1, Nn, hcat, 0, 128, 256);      // sum,mean,max

  // ---- sparse GIN layer 2 ----
  k_copy<<<gf(R1 * D), 256, 0, stream>>>(t2, x1, R1 * D);
  k_edge_scatter<<<gf(Etot * D), 256, 0, stream>>>(ei, Etot, x1, t2, D);
  nn_block(stream, t2, (int)R1, D, c[1], htmp, x2, bnm, bnr);
  k_pool<<<dim3(D, Bx), 256, 0, stream>>>(x2, Nn, hcat, 384, 512, 640);

  // ---- dense adjacency ----
  k_fill<<<gf((long long)Bx * Nn * Nn), 256, 0, stream>>>(adj, (long long)Bx * Nn * Nn, 0.f);
  k_adj_scatter<<<gf(Etot), 256, 0, stream>>>(ei, Etot, adj);

  // ---- mincut pool 1 (K1 = 274) ----
  launch_gemm(stream, x2, 0, D, 0, p1w, 0, K1, p1b, nullptr, 0, 0,
              s1, 0, K1, (int)R1, K1, D, 0, 1);
  k_softmax<<<(int)R1, 256, 0, stream>>>(s1, K1);
  launch_gemm(stream, s1, (long long)Nn * K1, K1, 1, x2, (long long)Nn * D, D,
              nullptr, nullptr, 0, 0, xp1, (long long)K1 * D, D, K1, D, Nn, 0, Bx);
  launch_gemm(stream, s1, (long long)Nn * K1, K1, 1, adj, (long long)Nn * Nn, Nn,
              nullptr, nullptr, 0, 0, sta1, (long long)K1 * Nn, Nn, K1, Nn, Nn, 0, Bx);
  launch_gemm(stream, sta1, (long long)K1 * Nn, Nn, 0, s1, (long long)Nn * K1, K1,
              nullptr, nullptr, 0, 0, oadj1, (long long)K1 * K1, K1, K1, K1, Nn, 0, Bx);
  k_trace<<<Bx, 256, 0, stream>>>(oadj1, K1, num1);
  k_rowsum<<<(int)R1, 256, 0, stream>>>(adj, Nn, deg1, 0);
  k_den<<<Bx, 256, 0, stream>>>(s1, deg1, Nn, K1, den1);
  launch_gemm(stream, s1, (long long)Nn * K1, K1, 1, s1, (long long)Nn * K1, K1,
              nullptr, nullptr, 0, 0, sta1, (long long)K1 * K1, K1, K1, K1, Nn, 0, Bx);
  k_ortho<<<Bx, 256, 0, stream>>>(sta1, K1, o1);
  k_zero_diag<<<gf((long long)Bx * K1), 256, 0, stream>>>(oadj1, K1);
  k_rowsum<<<(int)R2, 256, 0, stream>>>(oadj1, K1, dvec1, 1);
  k_adj_norm<<<gf((long long)Bx * K1 * K1), 256, 0, stream>>>(oadj1, dvec1, K1);
  k_pool<<<dim3(D, Bx), 256, 0, stream>>>(xp1, K1, hcat, 896, 768, 1024);  // mean,sum,max

  // ---- dense GIN c3 ----
  launch_gemm(stream, oadj1, (long long)K1 * K1, K1, 0, xp1, (long long)K1 * D, D,
              nullptr, xp1, (long long)K1 * D, D, tsh, (long long)K1 * D, D, K1, D, K1, 0, Bx);
  nn_block(stream, tsh, (int)R2, D, c[2], htmp, x3, bnm, bnr);
  k_pool<<<dim3(D, Bx), 256, 0, stream>>>(x3, K1, hcat, 1280, 1152, 1408);

  // ---- dense GIN c4 ----
  launch_gemm(stream, oadj1, (long long)K1 * K1, K1, 0, x3, (long long)K1 * D, D,
              nullptr, x3, (long long)K1 * D, D, tsh, (long long)K1 * D, D, K1, D, K1, 0, Bx);
  nn_block(stream, tsh, (int)R2, D, c[3], htmp, x4, bnm, bnr);
  k_pool<<<dim3(D, Bx), 256, 0, stream>>>(x4, K1, hcat, 1664, 1536, 1792);

  // ---- mincut pool 2 (K2 = 137) ----
  launch_gemm(stream, x4, 0, D, 0, p2w, 0, K2, p2b, nullptr, 0, 0,
              s2, 0, K2, (int)R2, K2, D, 0, 1);
  k_softmax<<<(int)R2, 256, 0, stream>>>(s2, K2);
  launch_gemm(stream, s2, (long long)K1 * K2, K2, 1, x4, (long long)K1 * D, D,
              nullptr, nullptr, 0, 0, xp2, (long long)K2 * D, D, K2, D, K1, 0, Bx);
  launch_gemm(stream, s2, (long long)K1 * K2, K2, 1, oadj1, (long long)K1 * K1, K1,
              nullptr, nullptr, 0, 0, sta2, (long long)K2 * K1, K1, K2, K1, K1, 0, Bx);
  launch_gemm(stream, sta2, (long long)K2 * K1, K1, 0, s2, (long long)K1 * K2, K2,
              nullptr, nullptr, 0, 0, oadj2, (long long)K2 * K2, K2, K2, K2, K1, 0, Bx);
  k_trace<<<Bx, 256, 0, stream>>>(oadj2, K2, num2);
  k_rowsum<<<(int)R2, 256, 0, stream>>>(oadj1, K1, deg2, 0);
  k_den<<<Bx, 256, 0, stream>>>(s2, deg2, K1, K2, den2);
  launch_gemm(stream, s2, (long long)K1 * K2, K2, 1, s2, (long long)K1 * K2, K2,
              nullptr, nullptr, 0, 0, sta2, (long long)K2 * K2, K2, K2, K2, K1, 0, Bx);
  k_ortho<<<Bx, 256, 0, stream>>>(sta2, K2, o2);
  k_zero_diag<<<gf((long long)Bx * K2), 256, 0, stream>>>(oadj2, K2);
  k_rowsum<<<(int)R3, 256, 0, stream>>>(oadj2, K2, dvec2, 1);
  k_adj_norm<<<gf((long long)Bx * K2 * K2), 256, 0, stream>>>(oadj2, dvec2, K2);
  k_pool<<<dim3(D, Bx), 256, 0, stream>>>(xp2, K2, hcat, 2048, 1920, 2176);

  // ---- dense GIN c5 ----
  launch_gemm(stream, oadj2, (long long)K2 * K2, K2, 0, xp2, (long long)K2 * D, D,
              nullptr, xp2, (long long)K2 * D, D, tsh, (long long)K2 * D, D, K2, D, K2, 0, Bx);
  nn_block(stream, tsh, (int)R3, D, c[4], htmp, x5, bnm, bnr);
  k_pool<<<dim3(D, Bx), 256, 0, stream>>>(x5, K2, hcat, 2432, 2304, 2560);

  // ---- head: Linear -> BN -> ReLU -> Linear ----
  launch_gemm(stream, hcat, 0, Fd, 0, f1w, 0, Hd, f1b, nullptr, 0, 0,
              h1, 0, Hd, Bx, Hd, Fd, 0, 1);
  k_bn_stats<<<Hd, 256, 0, stream>>>(h1, Bx, Hd, bnm, bnr);
  k_bn_relu<<<gf((long long)Bx * Hd), 256, 0, stream>>>(h1, bnm, bnr, f1g, f1be,
                                                        (long long)Bx * Hd, Hd);
  launch_gemm(stream, h1, 0, Hd, 0, f2w, 0, 4, f2b, nullptr, 0, 0,
              out, 0, 4, Bx, 4, Hd, 0, 1);

  // ---- scalar losses ----
  k_losses<<<1, 128, 0, stream>>>(num1, den1, o1, num2, den2, o2, out);
}
